// Attention_75935021793793
// MI455X (gfx1250) — matrix operand order repaired and checked by reading, not compiled, not run on hardware
//
#include <hip/hip_runtime.h>
#include <hip/hip_bf16.h>

typedef __bf16 bf16_t;
typedef __bf16 v16bf __attribute__((ext_vector_type(16)));
typedef __bf16 v8bf  __attribute__((ext_vector_type(8)));
typedef float  v8f   __attribute__((ext_vector_type(8)));
typedef int    v4i   __attribute__((ext_vector_type(4)));

#define S_LEN 2048
#define EMB   4096
#define HD    128
#define NH    32
#define NKV   8

#define AS1 __attribute__((address_space(1)))
#define AS3 __attribute__((address_space(3)))

#if defined(__has_builtin)
# if __has_builtin(__builtin_amdgcn_global_load_async_to_lds_b128)
#  define HAVE_ASYNC_LDS 1
# endif
#endif
#ifndef HAVE_ASYNC_LDS
# define HAVE_ASYNC_LDS 0
#endif

#if HAVE_ASYNC_LDS
static __device__ inline void async_copy16(const bf16_t* g, bf16_t* l) {
  __builtin_amdgcn_global_load_async_to_lds_b128(
      (AS1 v4i*)(unsigned long long)(const void*)g,
      (AS3 v4i*)(unsigned)(unsigned long long)(void*)l, 0, 0);
}
#endif

static __device__ inline void wait_async0() {
#if HAVE_ASYNC_LDS
# if __has_builtin(__builtin_amdgcn_s_wait_asynccnt)
  __builtin_amdgcn_s_wait_asynccnt(0);
  asm volatile("" ::: "memory");
# else
  asm volatile("s_wait_asynccnt 0" ::: "memory");
# endif
#endif
}

static __device__ inline v8f zero8() {
  v8f z;
#pragma unroll
  for (int i = 0; i < 8; ++i) z[i] = 0.0f;
  return z;
}

static __device__ inline v8f wmma_bf16(v16bf a, v16bf b, v8f c) {
  return __builtin_amdgcn_wmma_f32_16x16x32_bf16(false, a, false, b, (short)0, c, false, false);
}

static __device__ inline v16bf pack16(v8bf lo, v8bf hi) {
  v16bf r;
#pragma unroll
  for (int j = 0; j < 8; ++j) { r[j] = lo[j]; r[j + 8] = hi[j]; }
  return r;
}

// ---- A operand (16x32 bf16): lane l -> row m=l&15, 16 contiguous K at (l>>4)*16
static __device__ inline v16bf loadA_f32(const float* base, int ld, int row0, int k0, int lane) {
  const int m = lane & 15, hi = lane >> 4;
  const float* p = base + (long)(row0 + m) * ld + k0 + hi * 16;
  v16bf r;
#pragma unroll
  for (int j = 0; j < 8; ++j) { r[j] = (bf16_t)p[j]; r[j + 8] = (bf16_t)p[j + 8]; }
  return r;
}

static __device__ inline v16bf loadA_bf(const bf16_t* base, int ld, int row0, int k0, int lane) {
  const int m = lane & 15, hi = lane >> 4;
  const bf16_t* p = base + (long)(row0 + m) * ld + k0 + hi * 16;
  return pack16(*(const v8bf*)p, *(const v8bf*)(p + 8));
}

// ---- B operand (32x16), B = Mrows^T: lane l -> col n=l&15, 16 contiguous K at kb
static __device__ inline v16bf loadBT_f32(const float* Mrows, int ld, int n0, int k0, int lane) {
  const int n = lane & 15, kb = (lane >> 4) * 16;
  const float* p = Mrows + (long)(n0 + n) * ld + k0 + kb;
  v16bf r;
#pragma unroll
  for (int j = 0; j < 16; ++j) r[j] = (bf16_t)p[j];
  return r;
}

// ===========================================================================
// Epilogues for QKV: RoPE(Q,K) row-major store; V stored transposed.
// ===========================================================================
static __device__ inline void epilogue_qk(v8f a, int m0t, int colbase, bool isQ,
                                          const float* cosp, const float* sinp,
                                          bf16_t* qr, bf16_t* kr, int lane) {
  const int n = lane & 15, hi = lane >> 4;
  const int col = colbase + n;
  const int head = col >> 7, hd = col & 127, fi = hd >> 1;
  const bool evn = (hd & 1) == 0;
  bf16_t* dst = (isQ ? qr : kr) + (long)head * S_LEN * HD + hd;
#pragma unroll
  for (int r = 0; r < 8; ++r) {
    const int s = m0t + hi * 8 + r;
    float self  = a[r];
    float other = __shfl_xor(self, 1, 32);   // RoPE partner column in lane^1
    float c  = cosp[s * (HD / 2) + fi];
    float sn = sinp[s * (HD / 2) + fi];
    float o  = evn ? (self * c - other * sn) : (other * sn + self * c);
    dst[(long)s * HD] = (bf16_t)o;
  }
}

static __device__ inline void epilogue_v(v8f a, int m0t, int colbase, bf16_t* vT, int lane) {
  const int n = lane & 15, hi = lane >> 4;
  const int col = colbase + n;
  const int kvh = col >> 7, hd = col & 127;
  v8bf pv;
#pragma unroll
  for (int r = 0; r < 8; ++r) pv[r] = (bf16_t)a[r];
  *(v8bf*)(vT + (long)(kvh * HD + hd) * S_LEN + m0t + hi * 8) = pv;
}

// ===========================================================================
// Kernel 1: fused QKV projection + RoPE. Wave tile = 32x64 (2x4 WMMA tiles),
// A reused 4x, B reused 2x per K-step. N = 4096(Q)+1024(K)+1024(V).
// ===========================================================================
__global__ __launch_bounds__(128) void qkv_rope_kernel(
    const float* __restrict__ x, const float* __restrict__ cosp, const float* __restrict__ sinp,
    const float* __restrict__ wq, const float* __restrict__ wk, const float* __restrict__ wv,
    bf16_t* __restrict__ qr, bf16_t* __restrict__ kr, bf16_t* __restrict__ vT)
{
  const int lane = threadIdx.x & 31;
  const int wave = threadIdx.x >> 5;
  const int m0 = blockIdx.y * 32;                       // 64 strips
  const int cn = (blockIdx.x * 4 + wave) * 64;          // 96 n-waves

  const float* W; int wr0; bool isQ = false, isV = false;
  if (cn < 4096)      { W = wq; wr0 = cn;        isQ = true; }
  else if (cn < 5120) { W = wk; wr0 = cn - 4096; }
  else                { W = wv; wr0 = cn - 5120; isV = true; }

  v8f acc[2][4];
#pragma unroll
  for (int mi = 0; mi < 2; ++mi)
#pragma unroll
    for (int j = 0; j < 4; ++j) acc[mi][j] = zero8();

#pragma unroll 1
  for (int k0 = 0; k0 < EMB; k0 += 32) {
    v16bf a0 = loadA_f32(x, EMB, m0,      k0, lane);
    v16bf a1 = loadA_f32(x, EMB, m0 + 16, k0, lane);
#pragma unroll
    for (int j = 0; j < 4; ++j) {
      v16bf b = loadBT_f32(W, EMB, wr0 + j * 16, k0, lane);
      acc[0][j] = wmma_bf16(a0, b, acc[0][j]);
      acc[1][j] = wmma_bf16(a1, b, acc[1][j]);
    }
  }

#pragma unroll
  for (int mi = 0; mi < 2; ++mi)
#pragma unroll
    for (int j = 0; j < 4; ++j) {
      if (isV) epilogue_v(acc[mi][j], m0 + mi * 16, wr0 + j * 16, vT, lane);
      else     epilogue_qk(acc[mi][j], m0 + mi * 16, wr0 + j * 16, isQ, cosp, sinp, qr, kr, lane);
    }
}

// ===========================================================================
// K/V block staging into LDS (async DMA path, double-buffered).
// K block: rows k0..k0+31 x 128 -> dK[32][128]; V block: [128][k0..k0+31] -> dV[128][32]
// ===========================================================================
static __device__ inline void stage_kv(const bf16_t* K, const bf16_t* V, int k0,
                                       bf16_t* dK, bf16_t* dV, int lane) {
  const int hi = lane >> 4;
#if HAVE_ASYNC_LDS
#pragma unroll
  for (int j = 0; j < 16; ++j) {
    int row = k0 + 2 * j + hi;
    if (row > S_LEN - 1) row = S_LEN - 1;               // OOB keys masked later
    async_copy16(K + (long)row * HD + (lane & 15) * 8,
                 dK + (2 * j + hi) * HD + (lane & 15) * 8);
  }
#pragma unroll
  for (int j = 0; j < 16; ++j) {
    const int row = j * 8 + (lane >> 2);                // hd row, always valid
    int cc = k0 + (lane & 3) * 8;
    if (cc > S_LEN - 8) cc = S_LEN - 8;                 // OOB cols -> P==0 anyway
    async_copy16(V + (long)row * S_LEN + cc,
                 dV + row * 32 + (lane & 3) * 8);
  }
#else
#pragma unroll
  for (int j = 0; j < 16; ++j) {
    int row = k0 + 2 * j + hi;
    if (row > S_LEN - 1) row = S_LEN - 1;
    *(v8bf*)(dK + (2 * j + hi) * HD + (lane & 15) * 8) =
        *(const v8bf*)(K + (long)row * HD + (lane & 15) * 8);
  }
#pragma unroll
  for (int j = 0; j < 16; ++j) {
    const int row = j * 8 + (lane >> 2);
    int cc = k0 + (lane & 3) * 8;
    if (cc > S_LEN - 8) cc = S_LEN - 8;
    *(v8bf*)(dV + row * 32 + (lane & 3) * 8) = *(const v8bf*)(V + (long)row * S_LEN + cc);
  }
#endif
}

// ===========================================================================
// Kernel 2: flash-style causal attention, 1 wave/block, per (head, 16-q tile).
// Async double-buffered K/V in LDS; 8 score + 8 P*V WMMAs per 32-key block.
// ===========================================================================
__global__ __launch_bounds__(32) void flash_attn_kernel(
    const bf16_t* __restrict__ qr, const bf16_t* __restrict__ kr,
    const bf16_t* __restrict__ vT, bf16_t* __restrict__ oattn)
{
  __shared__ bf16_t ldsK[2][32 * HD];   // 2 x 8 KB
  __shared__ bf16_t ldsV[2][HD * 32];   // 2 x 8 KB
  __shared__ bf16_t ldsP[16 * 32];      // 1 KB

  const int lane = threadIdx.x & 31;
  const int qt = blockIdx.x & 127;
  const int h  = blockIdx.x >> 7;
  const int hkv = h >> 2;               // N_REP = 4
  const int q0 = qt * 16;
  const int n = lane & 15, hi = lane >> 4;
  const int kb = hi * 16;
  const float scale = 0.08838834764831845f;   // 1/sqrt(128)

  const bf16_t* Q = qr + (long)h   * S_LEN * HD;
  const bf16_t* K = kr + (long)hkv * S_LEN * HD;
  const bf16_t* V = vT + (long)hkv * HD * S_LEN;

  v16bf aq[4];
#pragma unroll
  for (int c = 0; c < 4; ++c) aq[c] = loadA_bf(Q, HD, q0, c * 32, lane);

  v8f acc[8];
#pragma unroll
  for (int t = 0; t < 8; ++t) acc[t] = zero8();
  float mrow[8], lrow[8];
#pragma unroll
  for (int r = 0; r < 8; ++r) { mrow[r] = -3.0e38f; lrow[r] = 0.0f; }

  const int nblk = (q0 + 47) >> 5;      // 32-key blocks covering keys 0..q0+15
  stage_kv(K, V, 0, ldsK[0], ldsV[0], lane);

  for (int blk = 0; blk < nblk; ++blk) {
    const int k0 = blk * 32;
    const int cur = blk & 1;
    const bf16_t* Kt = ldsK[cur];
    const bf16_t* Vt = ldsV[cur];
    wait_async0();                      // current buffers resident

    v8f s0 = zero8(), s1 = zero8();
#pragma unroll
    for (int c = 0; c < 4; ++c) {       // B[k][n] = Kt[sub*16+n][c*32+kb+j]
      const bf16_t* p0 = Kt + n * HD + c * 32 + kb;
      s0 = wmma_bf16(aq[c], pack16(*(const v8bf*)p0, *(const v8bf*)(p0 + 8)), s0);
    }
#pragma unroll
    for (int c = 0; c < 4; ++c) {
      const bf16_t* p1 = Kt + (16 + n) * HD + c * 32 + kb;
      s1 = wmma_bf16(aq[c], pack16(*(const v8bf*)p1, *(const v8bf*)(p1 + 8)), s1);
    }

    asm volatile("s_wait_dscnt 0" ::: "memory");  // K reads done; P WAR guard
    if (blk + 1 < nblk)                           // overlap next DMA with softmax+PV
      stage_kv(K, V, k0 + 32, ldsK[cur ^ 1], ldsV[cur ^ 1], lane);

    const bool full = (k0 + 31 <= q0);
    float rsc[8];
#pragma unroll
    for (int r = 0; r < 8; ++r) {
      float e0 = s0[r] * scale;
      float e1 = s1[r] * scale;
      if (!full) {
        const int qrow = q0 + hi * 8 + r;
        if (k0 + n      > qrow) e0 = -1.0e30f;
        if (k0 + 16 + n > qrow) e1 = -1.0e30f;
      }
      float mb = fmaxf(e0, e1);
#pragma unroll
      for (int off = 1; off < 16; off <<= 1) mb = fmaxf(mb, __shfl_xor(mb, off, 32));
      const float mnew = fmaxf(mrow[r], mb);
      const float sc = __expf(mrow[r] - mnew);
      const float p0 = __expf(e0 - mnew);
      const float p1 = __expf(e1 - mnew);
      float ps = p0 + p1;
#pragma unroll
      for (int off = 1; off < 16; off <<= 1) ps += __shfl_xor(ps, off, 32);
      lrow[r] = lrow[r] * sc + ps;
      mrow[r] = mnew;
      rsc[r] = sc;
      ldsP[(hi * 8 + r) * 32 + n]      = (bf16_t)p0;   // C-layout -> row-major 16x32
      ldsP[(hi * 8 + r) * 32 + 16 + n] = (bf16_t)p1;
    }
#pragma unroll
    for (int t = 0; t < 8; ++t)
#pragma unroll
      for (int r = 0; r < 8; ++r) acc[t][r] *= rsc[r];

    asm volatile("s_wait_dscnt 0" ::: "memory");       // P visible (DS in-order)
    const bf16_t* pa = ldsP + (lane & 15) * 32 + hi * 16;
    v16bf ap = pack16(*(const v8bf*)pa, *(const v8bf*)(pa + 8));  // A layout

#pragma unroll
    for (int t = 0; t < 8; ++t) {       // B[k][n] = Vt[t*16+n][kb+j]
      const bf16_t* pB = Vt + (t * 16 + n) * 32 + kb;
      acc[t] = wmma_bf16(ap, pack16(*(const v8bf*)pB, *(const v8bf*)(pB + 8)), acc[t]);
    }
  }

#pragma unroll
  for (int r = 0; r < 8; ++r) lrow[r] = 1.0f / lrow[r];
  bf16_t* ob = oattn + (long)q0 * (NH * HD) + h * HD + n;
#pragma unroll
  for (int t = 0; t < 8; ++t)
#pragma unroll
    for (int r = 0; r < 8; ++r)
      ob[(long)(hi * 8 + r) * (NH * HD) + t * 16] = (bf16_t)(acc[t][r] * lrow[r]);
}

// ===========================================================================
// Kernel 3: output projection, wave tile 32x64 (2x4 WMMA tiles), fp32 out.
// ===========================================================================
__global__ __launch_bounds__(128) void oproj_kernel(
    const bf16_t* __restrict__ oa, const float* __restrict__ wo, float* __restrict__ out)
{
  const int lane = threadIdx.x & 31;
  const int wave = threadIdx.x >> 5;
  const int m0 = blockIdx.y * 32;
  const int cn = (blockIdx.x * 4 + wave) * 64;

  v8f acc[2][4];
#pragma unroll
  for (int mi = 0; mi < 2; ++mi)
#pragma unroll
    for (int j = 0; j < 4; ++j) acc[mi][j] = zero8();

#pragma unroll 1
  for (int k0 = 0; k0 < EMB; k0 += 32) {
    v16bf a0 = loadA_bf(oa, EMB, m0,      k0, lane);
    v16bf a1 = loadA_bf(oa, EMB, m0 + 16, k0, lane);
#pragma unroll
    for (int j = 0; j < 4; ++j) {
      v16bf b = loadBT_f32(wo, EMB, cn + j * 16, k0, lane);
      acc[0][j] = wmma_bf16(a0, b, acc[0][j]);
      acc[1][j] = wmma_bf16(a1, b, acc[1][j]);
    }
  }

  const int n = lane & 15, hi = lane >> 4;
#pragma unroll
  for (int mi = 0; mi < 2; ++mi)
#pragma unroll
    for (int j = 0; j < 4; ++j) {
      float* dst = out + (long)(m0 + mi * 16) * EMB + cn + j * 16 + n;
#pragma unroll
      for (int r = 0; r < 8; ++r) dst[(long)(hi * 8 + r) * EMB] = acc[mi][j][r];
    }
}

// ===========================================================================
extern "C" void kernel_launch(void* const* d_in, const int* in_sizes, int n_in,
                              void* d_out, int out_size, void* d_ws, size_t ws_size,
                              hipStream_t stream) {
  (void)in_sizes; (void)n_in; (void)out_size; (void)ws_size;
  const float* x    = (const float*)d_in[0];
  const float* cosp = (const float*)d_in[1];
  const float* sinp = (const float*)d_in[2];
  const float* wq   = (const float*)d_in[3];
  const float* wk   = (const float*)d_in[4];
  const float* wv   = (const float*)d_in[5];
  const float* wo   = (const float*)d_in[6];

  bf16_t* qr = (bf16_t*)d_ws;                         //  32*2048*128 bf16 = 16 MB
  bf16_t* kr = qr + (size_t)NH  * S_LEN * HD;         //   8*2048*128 bf16 =  4 MB
  bf16_t* vT = kr + (size_t)NKV * S_LEN * HD;         //   8*128*2048 bf16 =  4 MB
  bf16_t* oa = vT + (size_t)NKV * HD * S_LEN;         //  2048*4096   bf16 = 16 MB

  qkv_rope_kernel<<<dim3(24, 64), 128, 0, stream>>>(x, cosp, sinp, wq, wk, wv, qr, kr, vT);
  flash_attn_kernel<<<dim3(4096), 32, 0, stream>>>(qr, kr, vT, oa);
  oproj_kernel<<<dim3(16, 64), 128, 0, stream>>>(oa, wo, (float*)d_out);
}
